// Model_82197084111504
// MI455X (gfx1250) — matrix operand order repaired and checked by reading, not compiled, not run on hardware
//
#include <hip/hip_runtime.h>

#define B_ 64
#define T_ 512
#define D_ 512
#define E_ 6
#define P_ 96
#define H_ 4
#define HD_ 128
#define FF_ 2048
#define BT_ (B_*T_)

typedef __attribute__((ext_vector_type(16))) __bf16 v16bf;
typedef __attribute__((ext_vector_type(8)))  float v8f;

__device__ __forceinline__ unsigned short f2bf(float f) {
  unsigned int u = __float_as_uint(f);
  u += 0x7FFFu + ((u >> 16) & 1u);           // round-to-nearest-even
  return (unsigned short)(u >> 16);
}
__device__ __forceinline__ float sigf(float x) { return 1.f / (1.f + __expf(-x)); }
// A-layout K mapping (ISA 7.12.2, 16-bit A 16x32): lane group g, element j -> kk
__device__ __forceinline__ int amap(int g, int j) { return (j < 8 ? j : j + 8) + g * 8; }

__device__ __forceinline__ v8f wmma_bf16(v16bf a, v16bf b, v8f c) {
  return __builtin_amdgcn_wmma_f32_16x16x32_bf16(false, a, false, b, (short)0, c, false, false);
}
__device__ __forceinline__ v16bf ldv(const unsigned short* p) {
  return *reinterpret_cast<const v16bf*>(p);
}

// ---------------- GARCH volatility scan (sequential, tiny) ----------------
__global__ void k_garch(const float* __restrict__ x_enc, const float* om_p,
                        const float* al_p, const float* be_p,
                        float* __restrict__ vol_last) {
  int b = threadIdx.x;
  if (b >= B_) return;
  float om = log1pf(__expf(om_p[0]));
  float al = 0.2f * sigf(al_p[0]);
  float be = 0.8f * sigf(be_p[0]);
  const float* r = x_enc + (size_t)b * T_ * E_ + (E_ - 1);
  float s = 0.f, s2 = 0.f;
  for (int t = 0; t < T_; ++t) { float v = r[(size_t)t * E_]; s += v; s2 += v * v; }
  float mean = s / T_;
  float var  = (s2 - s * mean) / (T_ - 1);
  float h = var + 1e-6f;
  for (int t = 0; t < T_ - 1; ++t) { float v = r[(size_t)t * E_]; h = om + al * v * v + be * h; }
  vol_last[b] = sqrtf(h);
}

// ------------- pack fp32 (M,K) row-major -> WMMA A-layout bf16 ------------
// out: [mt][kt][lane32][16]
__global__ void k_pack_a(const float* __restrict__ X, unsigned short* __restrict__ Ap,
                         int Mt, int Kt, int K) {
  long tid = (long)blockIdx.x * blockDim.x + threadIdx.x;
  long total = (long)Mt * Kt * 32;
  if (tid >= total) return;
  int lane = (int)(tid & 31); long tile = tid >> 5;
  int kt = (int)(tile % Kt); int mt = (int)(tile / Kt);
  int cl = lane & 15, g = lane >> 4;
  int row = mt * 16 + cl;
  const float* src = X + (long)row * K + kt * 32;
  unsigned short* dst = Ap + tid * 16;
#pragma unroll
  for (int j = 0; j < 16; ++j) dst[j] = f2bf(src[amap(g, j)]);
}

// ---- pack weight W (N,K) row-major -> WMMA B-layout bf16 (B = W^T tile) ---
// out: [nt][kt][lane32][16];  B[kk][nn] = W[nn][kk]
__global__ void k_pack_b(const float* __restrict__ W, unsigned short* __restrict__ Bp,
                         int Nt, int Kt, int K) {
  long tid = (long)blockIdx.x * blockDim.x + threadIdx.x;
  long total = (long)Nt * Kt * 32;
  if (tid >= total) return;
  int lane = (int)(tid & 31); long tile = tid >> 5;
  int kt = (int)(tile % Kt); int nt = (int)(tile / Kt);
  int cl = lane & 15, g = lane >> 4;
  int n = nt * 16 + cl;
  const float* src = W + (long)n * K + kt * 32;
  unsigned short* dst = Bp + tid * 16;
#pragma unroll
  for (int j = 0; j < 16; ++j) dst[j] = f2bf(src[amap(g, j)]);
}

// --------------------------- main WMMA GEMM -------------------------------
// C = A(bf16 packed) x B(bf16 packed) + bias ; per-wave: one 16x32 C slab.
// flags: 1=relu, 2=emit bf16 packed-A output (K_next = N), 4=emit fp32 row-major
__global__ void __launch_bounds__(256) k_gemm(
    const unsigned short* __restrict__ Ap, const unsigned short* __restrict__ Bp,
    const float* __restrict__ bias, float* __restrict__ Cf,
    unsigned short* __restrict__ Cpk, int Mt, int Nt2, int Kt, int N, int flags) {
  __shared__ float sC[8][16][32];
  int wave = threadIdx.x >> 5, lane = threadIdx.x & 31;
  long total = (long)Mt * Nt2;
  long tile = (long)blockIdx.x * 8 + wave;
  bool act = tile < total;
  if (!act) tile = 0;
  int np = (int)(tile % Nt2); int mt = (int)(tile / Nt2);
  const unsigned short* ar = Ap + (long)mt * Kt * 512;
  const unsigned short* b0 = Bp + (long)(2 * np) * Kt * 512;
  const unsigned short* b1 = b0 + (long)Kt * 512;
  v8f c0 = {}; v8f c1 = {};
  for (int kt = 0; kt < Kt; ++kt) {
    v16bf a  = ldv(ar + (kt * 32 + lane) * 16);
    v16bf bb0 = ldv(b0 + (kt * 32 + lane) * 16);
    v16bf bb1 = ldv(b1 + (kt * 32 + lane) * 16);
    c0 = wmma_bf16(a, bb0, c0);
    c1 = wmma_bf16(a, bb1, c1);
  }
  int cl = lane & 15, g = lane >> 4;
  int n0 = np * 32 + cl;
  float bv0 = bias[n0], bv1 = bias[n0 + 16];
#pragma unroll
  for (int r = 0; r < 8; ++r) {
    float v0 = c0[r] + bv0, v1 = c1[r] + bv1;
    if (flags & 1) { v0 = fmaxf(v0, 0.f); v1 = fmaxf(v1, 0.f); }
    int row = mt * 16 + r + 8 * g;
    if ((flags & 4) && act) {
      Cf[(long)row * N + n0]      = v0;
      Cf[(long)row * N + n0 + 16] = v1;
    }
    sC[wave][r + 8 * g][cl]      = v0;
    sC[wave][r + 8 * g][cl + 16] = v1;
  }
  __syncthreads();
  if (flags & 2) {
    unsigned short* dst = Cpk + ((long)(mt * Nt2 + np) * 32 + lane) * 16;
#pragma unroll
    for (int j = 0; j < 16; ++j) {
      unsigned short hv = f2bf(sC[wave][cl][amap(g, j)]);
      if (act) dst[j] = hv;
    }
  }
}

// ------------- LSTM: recurrent WMMA GEMM, h-state resident in LDS ---------
// grid.x = 4 (16 batch rows each), 1024 threads: wave w owns d-tile w.
__global__ void __launch_bounds__(1024) k_lstm(
    const float* __restrict__ x_enc, const float* __restrict__ w_ih,
    const float* __restrict__ b_ih, const float* __restrict__ b_hh,
    const unsigned short* __restrict__ Whh, float* __restrict__ xout) {
  __shared__ __align__(32) unsigned short hA[16 * 32 * 16];   // 16KB packed A of h
  int wave = threadIdx.x >> 5, lane = threadIdx.x & 31;
  int cl = lane & 15, g = lane >> 4;
  int mt = blockIdx.x;          // batch tile
  int dt = wave;                // d column tile 0..31
  int dcol = dt * 16 + cl;
  for (int i = threadIdx.x; i < 16 * 32 * 16; i += 1024) hA[i] = 0;
  float creg[8], hnew[8];
#pragma unroll
  for (int r = 0; r < 8; ++r) creg[r] = 0.f;
  __syncthreads();
  for (int t = 0; t < T_; ++t) {
    v8f ai = {}, af = {}, ag = {}, ao = {};
    for (int kt = 0; kt < 16; ++kt) {
      v16bf a  = ldv(&hA[(kt * 32 + lane) * 16]);
      v16bf w0 = ldv(Whh + ((long)((dt)      * 16 + kt) * 32 + lane) * 16);
      v16bf w1 = ldv(Whh + ((long)((32 + dt) * 16 + kt) * 32 + lane) * 16);
      v16bf w2 = ldv(Whh + ((long)((64 + dt) * 16 + kt) * 32 + lane) * 16);
      v16bf w3 = ldv(Whh + ((long)((96 + dt) * 16 + kt) * 32 + lane) * 16);
      ai = wmma_bf16(a, w0, ai); af = wmma_bf16(a, w1, af);
      ag = wmma_bf16(a, w2, ag); ao = wmma_bf16(a, w3, ao);
    }
#pragma unroll
    for (int r = 0; r < 8; ++r) {
      int brow = mt * 16 + r + 8 * g;
      const float* xt = x_enc + ((long)brow * T_ + t) * E_;
      float x0 = xt[0], x1 = xt[1], x2 = xt[2], x3 = xt[3], x4 = xt[4], x5 = xt[5];
      auto dot = [&](int n) -> float {
        const float* w = w_ih + (long)n * E_;
        return x0 * w[0] + x1 * w[1] + x2 * w[2] + x3 * w[3] + x4 * w[4] + x5 * w[5]
               + b_ih[n] + b_hh[n];
      };
      float iv = ai[r] + dot(dcol);
      float fv = af[r] + dot(512 + dcol);
      float gv = ag[r] + dot(1024 + dcol);
      float ov = ao[r] + dot(1536 + dcol);
      float c  = sigf(fv) * creg[r] + sigf(iv) * tanhf(gv);
      float hv = sigf(ov) * tanhf(c);
      creg[r] = c;
      hnew[r] = hv;
      xout[((long)brow * T_ + t) * D_ + dcol] = hv;
    }
    __syncthreads();   // all reads of h_{t-1} done
    {
      int kt2 = dcol >> 5, kk = dcol & 31;
      int gi = (kk >> 3) & 1, ji = (kk & 7) + (kk >= 16 ? 8 : 0);
#pragma unroll
      for (int r = 0; r < 8; ++r)
        hA[(kt2 * 32 + (gi * 16 + r + 8 * g)) * 16 + ji] = f2bf(hnew[r]);
    }
    __syncthreads();   // h_t visible before next step
  }
}

// ------------- attention operand re-packs from fp32 qkv -------------------
__global__ void k_pack_q(const float* __restrict__ qkv, unsigned short* __restrict__ qA) {
  int bh = blockIdx.y, b = bh >> 2, h = bh & 3;
  int tid = blockIdx.x * blockDim.x + threadIdx.x;   // 0..4095
  int lane = tid & 31; int tile = tid >> 5;          // tile = qt*4 + kt
  int kt = tile & 3, qt = tile >> 2;
  int cl = lane & 15, g = lane >> 4;
  const float* src = qkv + ((long)(b * T_ + qt * 16 + cl)) * 1536 + h * HD_ + kt * 32;
  unsigned short* dst = qA + (((long)bh * 128 + tile) * 32 + lane) * 16;
#pragma unroll
  for (int j = 0; j < 16; ++j) dst[j] = f2bf(src[amap(g, j)]);
}
__global__ void k_pack_kt(const float* __restrict__ qkv, unsigned short* __restrict__ kB) {
  int bh = blockIdx.y, b = bh >> 2, h = bh & 3;
  int tid = blockIdx.x * blockDim.x + threadIdx.x;
  int lane = tid & 31; int tile = tid >> 5;          // tile = nt*4 + kt (nn=t, kk=hd)
  int kt = tile & 3, nt = tile >> 2;
  int cl = lane & 15, g = lane >> 4;
  const float* src = qkv + ((long)(b * T_ + nt * 16 + cl)) * 1536 + 512 + h * HD_ + kt * 32;
  unsigned short* dst = kB + (((long)bh * 128 + tile) * 32 + lane) * 16;
#pragma unroll
  for (int j = 0; j < 16; ++j) dst[j] = f2bf(src[amap(g, j)]);
}
__global__ void k_pack_v(const float* __restrict__ qkv, unsigned short* __restrict__ vB) {
  int bh = blockIdx.y, b = bh >> 2, h = bh & 3;
  int tid = blockIdx.x * blockDim.x + threadIdx.x;
  int lane = tid & 31; int tile = tid >> 5;          // tile = kt2*8 + nt (kk=t, nn=hd)
  int nt = tile & 7, kt2 = tile >> 3;
  int cl = lane & 15, g = lane >> 4;
  const float* src = qkv + ((long)(b * T_ + kt2 * 32)) * 1536 + 1024 + h * HD_ + nt * 16 + cl;
  unsigned short* dst = vB + (((long)bh * 128 + tile) * 32 + lane) * 16;
#pragma unroll
  for (int j = 0; j < 16; ++j) dst[j] = f2bf(src[(long)amap(g, j) * 1536]);
}

// -------- flash attention: one wave per (bh, q-tile), WMMA QK^T and PV ----
__global__ void __launch_bounds__(32) k_attn(
    const unsigned short* __restrict__ qA, const unsigned short* __restrict__ kB,
    const unsigned short* __restrict__ vB, unsigned short* __restrict__ ctxA) {
  __shared__ __align__(32) unsigned short smP[512];  // P tile in A-layout
  __shared__ float smO[16 * 128];
  int qt = blockIdx.x, bh = blockIdx.y;
  int lane = threadIdx.x, cl = lane & 15, g = lane >> 4;
  const unsigned short* qbase = qA + ((long)bh * 128 + qt * 4) * 512;
  v16bf q[4];
#pragma unroll
  for (int kk = 0; kk < 4; ++kk) q[kk] = ldv(qbase + (kk * 32 + lane) * 16);
  const unsigned short* kbh = kB + (long)bh * 128 * 512;
  const unsigned short* vbh = vB + (long)bh * 128 * 512;
  float mrow[8], lrow[8];
  v8f O[8]; v8f zero = {};
#pragma unroll
  for (int r = 0; r < 8; ++r) { mrow[r] = -3.0e38f; lrow[r] = 0.f; O[r] = zero; }
  const float scale = 0.0883883476f;   // 1/sqrt(128)
  for (int kb = 0; kb < 16; ++kb) {    // 32 keys per block
    v8f s0 = {}; v8f s1 = {};
#pragma unroll
    for (int kk = 0; kk < 4; ++kk) {
      v16bf k0 = ldv(kbh + (((2 * kb) * 4 + kk) * 32 + lane) * 16);
      v16bf k1 = ldv(kbh + (((2 * kb + 1) * 4 + kk) * 32 + lane) * 16);
      s0 = wmma_bf16(q[kk], k0, s0);
      s1 = wmma_bf16(q[kk], k1, s1);
    }
#pragma unroll
    for (int r = 0; r < 8; ++r) {
      float a0 = s0[r] * scale, a1 = s1[r] * scale;
      float mx = fmaxf(a0, a1);
      for (int m = 1; m < 16; m <<= 1) mx = fmaxf(mx, __shfl_xor(mx, m, 32));
      float mnew = fmaxf(mrow[r], mx);
      float corr = __expf(mrow[r] - mnew);
      float p0 = __expf(a0 - mnew), p1 = __expf(a1 - mnew);
      float ps = p0 + p1;
      for (int m = 1; m < 16; m <<= 1) ps += __shfl_xor(ps, m, 32);
      lrow[r] = lrow[r] * corr + ps;
      mrow[r] = mnew;
#pragma unroll
      for (int nt = 0; nt < 8; ++nt) O[nt][r] *= corr;
      int rit = r + 8 * g;
      int kk0 = cl, kk1 = 16 + cl;
      int g0 = (kk0 >> 3) & 1, j0 = (kk0 & 7) + (kk0 >= 16 ? 8 : 0);
      int g1 = (kk1 >> 3) & 1, j1 = (kk1 & 7) + (kk1 >= 16 ? 8 : 0);
      smP[(g0 * 16 + rit) * 16 + j0] = f2bf(p0);
      smP[(g1 * 16 + rit) * 16 + j1] = f2bf(p1);
    }
    __syncthreads();
    v16bf pa = ldv(&smP[lane * 16]);
#pragma unroll
    for (int nt = 0; nt < 8; ++nt) {
      v16bf vm = ldv(vbh + ((kb * 8 + nt) * 32 + lane) * 16);
      O[nt] = wmma_bf16(pa, vm, O[nt]);
    }
    __syncthreads();
  }
#pragma unroll
  for (int r = 0; r < 8; ++r) {
    float inv = 1.f / lrow[r];
    int rit = r + 8 * g;
#pragma unroll
    for (int nt = 0; nt < 8; ++nt) smO[rit * 128 + nt * 16 + cl] = O[nt][r] * inv;
  }
  __syncthreads();
  int b = bh >> 2, h = bh & 3;
  int mt = b * 32 + qt;
#pragma unroll
  for (int kt = 0; kt < 4; ++kt) {
    unsigned short* dst = ctxA + (((long)mt * 16 + (h * 4 + kt)) * 32 + lane) * 16;
#pragma unroll
    for (int j = 0; j < 16; ++j) dst[j] = f2bf(smO[cl * 128 + kt * 32 + amap(g, j)]);
  }
}

// -------------------- residual add + LayerNorm (row-wise) -----------------
__global__ void __launch_bounds__(256) k_ln(
    const float* __restrict__ xin, const float* __restrict__ delta,
    const float* __restrict__ gam, const float* __restrict__ bet,
    float* __restrict__ xout) {
  __shared__ float red[256];
  long row = blockIdx.x;
  int tid = threadIdx.x;
  float v0 = xin[row * D_ + tid]       + delta[row * D_ + tid];
  float v1 = xin[row * D_ + tid + 256] + delta[row * D_ + tid + 256];
  red[tid] = v0 + v1; __syncthreads();
  for (int s = 128; s > 0; s >>= 1) { if (tid < s) red[tid] += red[tid + s]; __syncthreads(); }
  float mu = red[0] * (1.f / D_); __syncthreads();
  float d0 = v0 - mu, d1 = v1 - mu;
  red[tid] = d0 * d0 + d1 * d1; __syncthreads();
  for (int s = 128; s > 0; s >>= 1) { if (tid < s) red[tid] += red[tid + s]; __syncthreads(); }
  float rstd = rsqrtf(red[0] * (1.f / D_) + 1e-5f);
  xout[row * D_ + tid]       = d0 * rstd * gam[tid]       + bet[tid];
  xout[row * D_ + tid + 256] = d1 * rstd * gam[tid + 256] + bet[tid + 256];
}

// ------------ head: ai_residual, gate MLP, garch base, combine ------------
__global__ void __launch_bounds__(256) k_head(
    const float* __restrict__ x, const float* __restrict__ x_enc,
    const float* __restrict__ vol_last,
    const float* gp_w, const float* gp_b,
    const float* rh_w, const float* rh_b,
    const float* g1_w, const float* g1_b,
    const float* g2_w, const float* g2_b,
    float* __restrict__ out) {
  __shared__ float hid[256];
  __shared__ float gin[7];
  int b = blockIdx.x, tid = threadIdx.x;
  if (tid < 6) gin[tid] = x_enc[((long)b * T_ + (T_ - 1)) * E_ + tid];
  else if (tid == 6) gin[6] = vol_last[b];
  __syncthreads();
  float acc = g1_b[tid];
  for (int e = 0; e < 7; ++e) acc += gin[e] * g1_w[tid * 7 + e];
  hid[tid] = fmaxf(acc, 0.f);
  __syncthreads();
  if (tid < P_) {
    float gacc = g2_b[tid];
    for (int k = 0; k < 256; ++k) gacc += hid[k] * g2_w[tid * 256 + k];
    float gate = sigf(gacc);
    const float* xl = x + ((long)b * T_ + (T_ - 1)) * D_;
    float ai = rh_b[tid];
    for (int k = 0; k < D_; ++k) ai += xl[k] * rh_w[(long)tid * D_ + k];
    float gb = vol_last[b] * gp_w[tid] + gp_b[tid];
    out[b * P_ + tid] = gb + gate * ai;
  }
}

extern "C" void kernel_launch(void* const* d_in, const int* in_sizes, int n_in,
                              void* d_out, int out_size, void* d_ws, size_t ws_size,
                              hipStream_t stream) {
  (void)in_sizes; (void)n_in; (void)out_size; (void)ws_size;
  const float* x_enc = (const float*)d_in[0];
  const float* omega = (const float*)d_in[4];
  const float* alpha = (const float*)d_in[5];
  const float* beta  = (const float*)d_in[6];
  const float* gp_w  = (const float*)d_in[7];
  const float* gp_b  = (const float*)d_in[8];
  const float* w_ih  = (const float*)d_in[9];
  const float* w_hh  = (const float*)d_in[10];
  const float* b_ih  = (const float*)d_in[11];
  const float* b_hh  = (const float*)d_in[12];
  const float* attn_in_w  = (const float*)d_in[13];
  const float* attn_in_b  = (const float*)d_in[14];
  const float* attn_out_w = (const float*)d_in[15];
  const float* attn_out_b = (const float*)d_in[16];
  const float* ln1_g = (const float*)d_in[17];
  const float* ln1_b = (const float*)d_in[18];
  const float* ffn_w1 = (const float*)d_in[19];
  const float* ffn_b1 = (const float*)d_in[20];
  const float* ffn_w2 = (const float*)d_in[21];
  const float* ffn_b2 = (const float*)d_in[22];
  const float* ln2_g = (const float*)d_in[23];
  const float* ln2_b = (const float*)d_in[24];
  const float* rh_w = (const float*)d_in[25];
  const float* rh_b = (const float*)d_in[26];
  const float* g1_w = (const float*)d_in[27];
  const float* g1_b = (const float*)d_in[28];
  const float* g2_w = (const float*)d_in[29];
  const float* g2_b = (const float*)d_in[30];
  float* out = (float*)d_out;

  char* ws = (char*)d_ws;
  size_t off = 0;
  auto take = [&](size_t bytes) -> char* {
    char* p = ws + off; off += (bytes + 255) & ~(size_t)255; return p;
  };
  float*          vol   = (float*)take(B_ * sizeof(float));
  float*          xbuf  = (float*)take((size_t)BT_ * D_ * 4);       // activations
  unsigned short* xA    = (unsigned short*)take((size_t)BT_ * D_ * 2);
  float*          big   = (float*)take((size_t)BT_ * 3 * D_ * 4);  // qkv / attn_out / ff
  unsigned short* qAb   = (unsigned short*)take((size_t)B_ * H_ * T_ * HD_ * 2);
  unsigned short* kBb   = (unsigned short*)take((size_t)B_ * H_ * T_ * HD_ * 2);
  unsigned short* vBb   = (unsigned short*)take((size_t)B_ * H_ * T_ * HD_ * 2);
  unsigned short* ctxA  = (unsigned short*)take((size_t)B_ * H_ * T_ * HD_ * 2);
  unsigned short* hAbuf = (unsigned short*)take((size_t)BT_ * FF_ * 2);
  unsigned short* wB    = (unsigned short*)take((size_t)4 << 20);

  k_garch<<<1, 64, 0, stream>>>(x_enc, omega, alpha, beta, vol);

  // LSTM: pack w_hh (2048x512) then recurrent WMMA
  k_pack_b<<<256, 256, 0, stream>>>(w_hh, wB, 128, 16, 512);
  k_lstm<<<4, 1024, 0, stream>>>(x_enc, w_ih, b_ih, b_hh, wB, xbuf);

  for (int l = 0; l < 2; ++l) {
    // qkv = x @ Win^T + b
    k_pack_a<<<4096, 256, 0, stream>>>(xbuf, xA, 2048, 16, 512);
    k_pack_b<<<192, 256, 0, stream>>>(attn_in_w + (size_t)l * 3 * D_ * D_, wB, 96, 16, 512);
    k_gemm<<<12288, 256, 0, stream>>>(xA, wB, attn_in_b + l * 3 * D_, big, nullptr,
                                      2048, 48, 16, 1536, 4);
    // attention operand packs + flash attention
    k_pack_q <<<dim3(16, 256), 256, 0, stream>>>(big, qAb);
    k_pack_kt<<<dim3(16, 256), 256, 0, stream>>>(big, kBb);
    k_pack_v <<<dim3(16, 256), 256, 0, stream>>>(big, vBb);
    k_attn<<<dim3(32, 256), 32, 0, stream>>>(qAb, kBb, vBb, ctxA);
    // out projection
    k_pack_b<<<64, 256, 0, stream>>>(attn_out_w + (size_t)l * D_ * D_, wB, 32, 16, 512);
    k_gemm<<<4096, 256, 0, stream>>>(ctxA, wB, attn_out_b + l * D_, big, nullptr,
                                     2048, 16, 16, 512, 4);
    k_ln<<<BT_, 256, 0, stream>>>(xbuf, big, ln1_g + l * D_, ln1_b + l * D_, xbuf);
    // FFN (relu output emitted directly as packed-A bf16)
    k_pack_a<<<4096, 256, 0, stream>>>(xbuf, xA, 2048, 16, 512);
    k_pack_b<<<256, 256, 0, stream>>>(ffn_w1 + (size_t)l * FF_ * D_, wB, 128, 16, 512);
    k_gemm<<<16384, 256, 0, stream>>>(xA, wB, ffn_b1 + l * FF_, nullptr, hAbuf,
                                      2048, 64, 16, 2048, 1 | 2);
    k_pack_b<<<256, 256, 0, stream>>>(ffn_w2 + (size_t)l * D_ * FF_, wB, 32, 64, 2048);
    k_gemm<<<2048, 256, 0, stream>>>(hAbuf, wB, ffn_b2 + l * D_, big, nullptr,
                                     2048, 8, 64, 512, 4);
    k_ln<<<BT_, 256, 0, stream>>>(xbuf, big, ln2_g + l * D_, ln2_b + l * D_, xbuf);
  }

  k_head<<<64, 256, 0, stream>>>(xbuf, x_enc, vol, gp_w, gp_b, rh_w, rh_b,
                                 g1_w, g1_b, g2_w, g2_b, out);
}
